// Net_63788854280300
// MI455X (gfx1250) — matrix operand order repaired and checked by reading, not compiled, not run on hardware
//
#include <hip/hip_runtime.h>
#include <hip/hip_bf16.h>
#include <math.h>

// GIN (3 layers, D=32) + pool + head for MI455X (gfx1250, wave32).
// Edge scatter = atomic f32 (L2-resident: working set ~38MB << 192MB L2).
// Node MLPs = v_wmma_f32_16x16x32_f16: K=32 == hidden dim, one wave per
// 16-node tile, 2 WMMAs per GEMM (output column halves), f32 accumulate.
// Weight B-fragments are pre-swizzled to f16 once per launch so the hot
// kernels load them with 2 coalesced global_load_b128 per fragment.

typedef _Float16 v16h __attribute__((ext_vector_type(16)));
typedef float    v8f  __attribute__((ext_vector_type(8)));

#define D    32
#define FIN  3
#define FRAG_HALFS 1024  // per matrix: 2 colhalves * 32 lanes * 16 halfs

static __device__ __forceinline__ v8f wmma_f16(v16h a, v16h b, v8f c) {
  // (neg_a, A, neg_b, B, c_mod, C, reuse_a, reuse_b)
  return __builtin_amdgcn_wmma_f32_16x16x32_f16(false, a, false, b, (short)0, c,
                                                false, false);
}

// A-fragment (16x32 f16) per CDNA5 ISA layout:
//   lanes 0-15: row M = lane, K in {0..7} U {16..23}  (VGPR v -> K {2v,2v+1} / {16+2v,...})
//   lanes 16-31: row M = lane-16, same but K offset +8
static __device__ __forceinline__ v16h load_a_frag(const float* rowptr, int half) {
  v16h a;
#pragma unroll
  for (int v = 0; v < 8; ++v) {
    int k0 = ((v < 4) ? (2 * v) : (2 * (v - 4) + 16)) + 8 * half;
    a[2 * v]     = (_Float16)rowptr[k0];
    a[2 * v + 1] = (_Float16)rowptr[k0 + 1];
  }
  return a;
}

// ---------------- weight fragment prep (runs once per launch) ----------------
// B-fragment (32x16 f16), W row-major [kin x 32], output cols [colbase, colbase+16):
//   lane = column (lane&15); lanes 0-15 hold K=0..7,16..23, lanes 16-31 hold K=8..15,24..31,
//   two K per VGPR.  K >= kin zero-padded (layer 1 has kin=3).
// Output layout: [colhalf][lane][16 halfs] -> one v16h per (colhalf,lane).
__global__ __launch_bounds__(64) void k_prep_bfrag(const float* __restrict__ W,
                                                   int kin,
                                                   _Float16* __restrict__ outf) {
  const int t = threadIdx.x;            // 0..63
  const int ch = t >> 5;                // colhalf 0/1
  const int lane = t & 31;
  const int row16 = lane & 15;
  const int half = lane >> 4;
  const int colbase = ch << 4;
  _Float16* o = outf + ((size_t)(ch * 32 + lane)) * 16;
#pragma unroll
  for (int v = 0; v < 8; ++v) {
    int k = ((v < 4) ? (2 * v) : (2 * (v - 4) + 16)) + 8 * half;
    float w0 = (k     < kin) ? W[(size_t)k       * D + colbase + row16] : 0.f;
    float w1 = (k + 1 < kin) ? W[(size_t)(k + 1) * D + colbase + row16] : 0.f;
    o[2 * v]     = (_Float16)w0;
    o[2 * v + 1] = (_Float16)w1;
  }
}

// ---------------- elementwise / scatter kernels ----------------

__global__ void k_init_pad(const float* __restrict__ x, float* __restrict__ agg, int n) {
  int t = blockIdx.x * blockDim.x + threadIdx.x;
  if (t >= n * D) return;
  int node = t >> 5, c = t & 31;
  agg[t] = (c < FIN) ? x[(size_t)node * FIN + c] : 0.f;
}

__global__ void k_copy4(const float4* __restrict__ s, float4* __restrict__ d, int n4) {
  int t = blockIdx.x * blockDim.x + threadIdx.x;
  if (t < n4) d[t] = s[t];
}

__global__ void k_zero(float* __restrict__ p, int n) {
  int t = blockIdx.x * blockDim.x + threadIdx.x;
  if (t < n) p[t] = 0.f;
}

// layer-1 scatter: 3-wide features, thread = (edge, comp)
__global__ void k_scatter3(const float* __restrict__ x, const int* __restrict__ src,
                           const int* __restrict__ dst, float* __restrict__ agg, int E_) {
  int t = blockIdx.x * blockDim.x + threadIdx.x;
  if (t >= E_ * 4) return;
  int e = t >> 2, c = t & 3;
  if (c >= FIN) return;
  atomicAdd(&agg[((size_t)dst[e] << 5) + c], x[(size_t)src[e] * FIN + c]);
}

// 32-wide scatter: wave per edge, lane per channel -> coalesced 128B gather + atomic
__global__ void k_scatter32(const float* __restrict__ h, const int* __restrict__ src,
                            const int* __restrict__ dst, float* __restrict__ agg, int E_) {
  int t = blockIdx.x * blockDim.x + threadIdx.x;
  if (t >= (E_ << 5)) return;
  int e = t >> 5, c = t & 31;
  atomicAdd(&agg[((size_t)dst[e] << 5) + c], h[((size_t)src[e] << 5) + c]);
}

// global_add_pool: p[batch[n], c] += h[n, c]
__global__ void k_pool(const float* __restrict__ h, const int* __restrict__ batch,
                       float* __restrict__ p, int n) {
  int t = blockIdx.x * blockDim.x + threadIdx.x;
  if (t >= n * D) return;
  int node = t >> 5, c = t & 31;
  atomicAdd(&p[((size_t)batch[node] << 5) + c], h[t]);
}

// ---------------- WMMA GIN MLP: relu(agg@Wa+ba)@Wb+bb -> relu -> BN ----------------

__global__ __launch_bounds__(32) void k_gin_mlp(
    const float* __restrict__ agg, float* __restrict__ out,
    const _Float16* __restrict__ waf, const float* __restrict__ ba,
    const _Float16* __restrict__ wbf, const float* __restrict__ bbv,
    const float* __restrict__ gam, const float* __restrict__ bet,
    const float* __restrict__ mu, const float* __restrict__ var,
    int nrows) {
  __shared__ float lds[16 * D];  // inter-GEMM transpose bounce (2KB, one wave/block)
  const int lane = threadIdx.x;
  const int row16 = lane & 15;
  const int half = lane >> 4;
  const int rbase = blockIdx.x * 16;
  const int col0 = row16, col1 = row16 + 16;

  // pre-swizzled B fragments: one v16h (32B) per lane per colhalf
  const v16h* wa = (const v16h*)waf;
  const v16h* wb = (const v16h*)wbf;
  v16h bw0 = wa[lane];
  v16h bw1 = wa[32 + lane];

  // ---- GEMM1: T = agg_tile[16,32] @ Wa ----
  int arow = rbase + row16;
  if (arow >= nrows) arow = nrows - 1;  // clamp (dup row, stores guarded)
  v16h af = load_a_frag(agg + (size_t)arow * D, half);

  v8f c0 = {}, c1 = {};
  c0 = wmma_f16(af, bw0, c0);
  c1 = wmma_f16(af, bw1, c1);

  // epilogue1: +ba, relu, stage to LDS row-major [16][32]
  {
    float bia0 = ba[col0], bia1 = ba[col1];
#pragma unroll
    for (int r = 0; r < 8; ++r) {
      int m = (half ? 8 : 0) + r;  // D-frag: lanes 0-15 -> M=r, lanes 16-31 -> M=8+r
      float t0 = c0[r] + bia0; t0 = t0 > 0.f ? t0 : 0.f;
      float t1 = c1[r] + bia1; t1 = t1 > 0.f ? t1 : 0.f;
      lds[m * D + col0] = t0;
      lds[m * D + col1] = t1;
    }
  }
  __syncthreads();  // single-wave WG: barrier is a NOP, forces DS order/wait

  // ---- GEMM2: H2 = relu(T) @ Wb ----
  v16h a2 = load_a_frag(&lds[row16 * D], half);
  v16h bu0 = wb[lane];
  v16h bu1 = wb[32 + lane];
  v8f d0 = {}, d1 = {};
  d0 = wmma_f16(a2, bu0, d0);
  d1 = wmma_f16(a2, bu1, d1);

  // epilogue2: +bb, relu, BN folded to out = h2*s + (beta - mean*s)
  float s0 = gam[col0] * rsqrtf(var[col0] + 1e-5f);
  float s1 = gam[col1] * rsqrtf(var[col1] + 1e-5f);
  float o0 = bet[col0] - mu[col0] * s0;
  float o1 = bet[col1] - mu[col1] * s1;
  float bi0 = bbv[col0], bi1 = bbv[col1];
#pragma unroll
  for (int r = 0; r < 8; ++r) {
    int m = (half ? 8 : 0) + r;
    int row = rbase + m;
    if (row < nrows) {
      float t0 = d0[r] + bi0; t0 = t0 > 0.f ? t0 : 0.f;
      float t1 = d1[r] + bi1; t1 = t1 > 0.f ? t1 : 0.f;
      out[(size_t)row * D + col0] = t0 * s0 + o0;
      out[(size_t)row * D + col1] = t1 * s1 + o1;
    }
  }
}

// ---------------- head: tanh(relu(p@Wf1+bf1) @ Wf2 + bf2) ----------------

__global__ __launch_bounds__(32) void k_head(
    const float* __restrict__ p, const _Float16* __restrict__ wf1f,
    const float* __restrict__ bf1, const float* __restrict__ Wf2,
    const float* __restrict__ bf2, float* __restrict__ out, int G_) {
  __shared__ float lds[16 * D];
  const int lane = threadIdx.x, row16 = lane & 15, half = lane >> 4;
  const int gbase = blockIdx.x * 16;
  const int col0 = row16, col1 = row16 + 16;

  const v16h* wf = (const v16h*)wf1f;
  int arow = gbase + row16;
  if (arow >= G_) arow = G_ - 1;
  v16h a = load_a_frag(p + (size_t)arow * D, half);
  v16h b0 = wf[lane];
  v16h b1 = wf[32 + lane];
  v8f c0 = {}, c1 = {};
  c0 = wmma_f16(a, b0, c0);
  c1 = wmma_f16(a, b1, c1);

  float bia0 = bf1[col0], bia1 = bf1[col1];
#pragma unroll
  for (int r = 0; r < 8; ++r) {
    int m = (half ? 8 : 0) + r;
    float t0 = c0[r] + bia0; lds[m * D + col0] = t0 > 0.f ? t0 : 0.f;
    float t1 = c1[r] + bia1; lds[m * D + col1] = t1 > 0.f ? t1 : 0.f;
  }
  __syncthreads();

  if (lane < 16) {
    int g = gbase + lane;
    if (g < G_) {
      float acc = bf2[0];
#pragma unroll
      for (int k = 0; k < D; ++k) acc += lds[lane * D + k] * Wf2[k];
      out[g] = tanhf(acc);
    }
  }
}

// ---------------- launch ----------------

extern "C" void kernel_launch(void* const* d_in, const int* in_sizes, int n_in,
                              void* d_out, int out_size, void* d_ws, size_t ws_size,
                              hipStream_t stream) {
  (void)n_in; (void)ws_size;
  const float* x     = (const float*)d_in[0];
  const int*   ei    = (const int*)d_in[1];
  const int*   batch = (const int*)d_in[2];
  const int N = in_sizes[2];
  const int E = in_sizes[1] / 2;
  const int G = out_size;
  const int* srcp = ei;
  const int* dstp = ei + E;

  const float* W1a=(const float*)d_in[3];  const float* b1a=(const float*)d_in[4];
  const float* W1b=(const float*)d_in[5];  const float* b1b=(const float*)d_in[6];
  const float* g1 =(const float*)d_in[7];  const float* be1=(const float*)d_in[8];
  const float* m1 =(const float*)d_in[9];  const float* v1 =(const float*)d_in[10];
  const float* W2a=(const float*)d_in[11]; const float* b2a=(const float*)d_in[12];
  const float* W2b=(const float*)d_in[13]; const float* b2b=(const float*)d_in[14];
  const float* g2 =(const float*)d_in[15]; const float* be2=(const float*)d_in[16];
  const float* m2 =(const float*)d_in[17]; const float* v2 =(const float*)d_in[18];
  const float* W3a=(const float*)d_in[19]; const float* b3a=(const float*)d_in[20];
  const float* W3b=(const float*)d_in[21]; const float* b3b=(const float*)d_in[22];
  const float* g3 =(const float*)d_in[23]; const float* be3=(const float*)d_in[24];
  const float* m3 =(const float*)d_in[25]; const float* v3 =(const float*)d_in[26];
  const float* Wf1=(const float*)d_in[27]; const float* bf1=(const float*)d_in[28];
  const float* Wf2=(const float*)d_in[29]; const float* bf2=(const float*)d_in[30];

  float* ws   = (float*)d_ws;
  float* agg  = ws;                       // [N,32]
  float* hA   = ws + (size_t)N * D;       // [N,32]
  float* hB   = ws + 2 * (size_t)N * D;   // [N,32]
  float* pool = ws + 3 * (size_t)N * D;   // [G,32]
  // f16 weight fragments: 7 matrices * 1024 halfs (32B-aligned: offsets are
  // multiples of 128B given hipMalloc's 256B base alignment)
  _Float16* wfr = (_Float16*)(ws + 3 * (size_t)N * D + (size_t)G * D);
  _Float16* f1a = wfr + 0 * FRAG_HALFS;
  _Float16* f1b = wfr + 1 * FRAG_HALFS;
  _Float16* f2a = wfr + 2 * FRAG_HALFS;
  _Float16* f2b = wfr + 3 * FRAG_HALFS;
  _Float16* f3a = wfr + 4 * FRAG_HALFS;
  _Float16* f3b = wfr + 5 * FRAG_HALFS;
  _Float16* ff1 = wfr + 6 * FRAG_HALFS;
  float* outp = (float*)d_out;            // [G]

  const int TB = 256;
  const int ndGrid = (N * D + TB - 1) / TB;
  const int tiles = (N + 15) / 16;
  const int e32Grid = (int)(((long long)E * D + TB - 1) / TB);

  // weight fragment prep (tiny, once per launch)
  k_prep_bfrag<<<1, 64, 0, stream>>>(W1a, FIN, f1a);
  k_prep_bfrag<<<1, 64, 0, stream>>>(W1b, D,   f1b);
  k_prep_bfrag<<<1, 64, 0, stream>>>(W2a, D,   f2a);
  k_prep_bfrag<<<1, 64, 0, stream>>>(W2b, D,   f2b);
  k_prep_bfrag<<<1, 64, 0, stream>>>(W3a, D,   f3a);
  k_prep_bfrag<<<1, 64, 0, stream>>>(W3b, D,   f3b);
  k_prep_bfrag<<<1, 64, 0, stream>>>(Wf1, D,   ff1);

  // layer 1 (F_IN=3, zero-padded to K=32 in agg and in the Wa fragment)
  k_init_pad<<<ndGrid, TB, 0, stream>>>(x, agg, N);
  k_scatter3<<<(E * 4 + TB - 1) / TB, TB, 0, stream>>>(x, srcp, dstp, agg, E);
  k_gin_mlp<<<tiles, 32, 0, stream>>>(agg, hA, f1a, b1a, f1b, b1b, g1, be1, m1, v1, N);
  // layer 2
  k_copy4<<<(N * D / 4 + TB - 1) / TB, TB, 0, stream>>>((const float4*)hA, (float4*)agg, N * D / 4);
  k_scatter32<<<e32Grid, TB, 0, stream>>>(hA, srcp, dstp, agg, E);
  k_gin_mlp<<<tiles, 32, 0, stream>>>(agg, hB, f2a, b2a, f2b, b2b, g2, be2, m2, v2, N);
  // layer 3
  k_copy4<<<(N * D / 4 + TB - 1) / TB, TB, 0, stream>>>((const float4*)hB, (float4*)agg, N * D / 4);
  k_scatter32<<<e32Grid, TB, 0, stream>>>(hB, srcp, dstp, agg, E);
  k_gin_mlp<<<tiles, 32, 0, stream>>>(agg, hA, f3a, b3a, f3b, b3b, g3, be3, m3, v3, N);
  // pool + head
  k_zero<<<(G * D + TB - 1) / TB, TB, 0, stream>>>(pool, G * D);
  k_pool<<<ndGrid, TB, 0, stream>>>(hA, batch, pool, N);
  k_head<<<(G + 15) / 16, 32, 0, stream>>>(pool, ff1, bf1, Wf2, bf2, outp, G);
}
